// Model_21251498180718
// MI455X (gfx1250) — compile-verified, matrix operand order checked by reading, not run on hardware
//
#include <hip/hip_runtime.h>
#include <hip/hip_bf16.h>
#include <stdint.h>

// ---------------------------------------------------------------------------
// MLP forward for MI455X (gfx1250, wave32, WMMA):
//   h = relu(x @ W1 + b1); out = h @ W2 + b2
// fc1: 256x128 block tile, 8 waves, wave tile 64x64 (16 WMMAs / K-step),
//      double-buffered LDS filled by GLOBAL_LOAD_ASYNC_TO_LDS_B128 (ASYNCcnt),
//      overlapping the async DMA with WMMA compute. K-loop kept un-unrolled
//      so the wave stays under 256 VGPRs (no MSB bank shuffling moves).
// fc2: streaming WMMA over K=4096 with N padded to 16.
// ---------------------------------------------------------------------------

#define IN_DIM  3072
#define HIDDEN  4096
#define OUT_DIM 10
#define BATCH   8192

typedef __attribute__((ext_vector_type(16))) __bf16 v16bf;
typedef __attribute__((ext_vector_type(8)))  float  v8f;
typedef __attribute__((ext_vector_type(4)))  int    v4i;

union V16 { v16bf v; int4 i[2]; };
union PK4 { __bf16 h[4]; unsigned long long u; };

#if __has_builtin(__builtin_amdgcn_global_load_async_to_lds_b128) && \
    __has_builtin(__builtin_amdgcn_s_wait_asynccnt)
#define HAVE_ASYNC_LDS 1
#else
#define HAVE_ASYNC_LDS 0
#endif

// 16-byte global -> LDS copy. Async DMA path (ASYNCcnt) on CDNA5.
__device__ __forceinline__ void copy16_g2l(void* lds, const void* g) {
#if HAVE_ASYNC_LDS
  __builtin_amdgcn_global_load_async_to_lds_b128(
      (__attribute__((address_space(1))) v4i*)g,
      (__attribute__((address_space(3))) v4i*)lds,
      /*offset=*/0, /*cpol=*/0);
#else
  *(int4*)lds = *(const int4*)g;
#endif
}

__device__ __forceinline__ void wait_g2l() {
#if HAVE_ASYNC_LDS
  __builtin_amdgcn_s_wait_asynccnt(0);
#endif
}

// ---------------- prep kernels --------------------------------------------

__global__ void cvt_f32_bf16x4(const float* __restrict__ in,
                               __bf16* __restrict__ out, int n4) {
  int i = blockIdx.x * blockDim.x + threadIdx.x;
  if (i >= n4) return;
  float4 v = ((const float4*)in)[i];
  PK4 p;
  p.h[0] = (__bf16)v.x; p.h[1] = (__bf16)v.y;
  p.h[2] = (__bf16)v.z; p.h[3] = (__bf16)v.w;
  ((unsigned long long*)out)[i] = p.u;
}

// W1 [IN_DIM][HIDDEN] f32  ->  W1T [HIDDEN][IN_DIM] bf16 (coalesced both sides)
__global__ __launch_bounds__(256) void w1_transpose(const float* __restrict__ W1,
                                                    __bf16* __restrict__ W1T) {
  __shared__ __bf16 tile[32][33];
  const int tx = threadIdx.x & 31;
  const int ty = threadIdx.x >> 5;          // 0..7
  const int n0 = blockIdx.x * 32;
  const int k0 = blockIdx.y * 32;
#pragma unroll
  for (int r = 0; r < 4; ++r) {
    int kl = ty + r * 8;
    tile[kl][tx] = (__bf16)W1[(size_t)(k0 + kl) * HIDDEN + n0 + tx];
  }
  __syncthreads();
#pragma unroll
  for (int r = 0; r < 4; ++r) {
    int nl = ty + r * 8;
    W1T[(size_t)(n0 + nl) * IN_DIM + k0 + tx] = tile[tx][nl];
  }
}

// W2 [HIDDEN][10] f32 -> W2T [16][HIDDEN] bf16 (cols 10..15 zero-padded)
__global__ void w2_prep(const float* __restrict__ W2, __bf16* __restrict__ W2T) {
  int i = blockIdx.x * blockDim.x + threadIdx.x;   // 0 .. 16*HIDDEN-1
  int k = i & (HIDDEN - 1);
  int n = i >> 12;                                 // HIDDEN == 4096
  __bf16 v = (__bf16)0.0f;
  if (n < OUT_DIM) v = (__bf16)W2[(size_t)k * OUT_DIM + n];
  W2T[(size_t)n * HIDDEN + k] = v;
}

// ---------------- fc1: [8192,3072] x [3072,4096] + bias, relu -> bf16 -----
// Block tile 256(M) x 128(N), K-step 32. 8 waves: 4 along M x 2 along N,
// wave tile 64x64 = 4x4 WMMA tiles. LDS rows padded to 40 halves (80B) so
// the 16B fragment loads are bank-conflict-free. Double-buffered LDS.

__global__ __launch_bounds__(256) void fc1_gemm(const __bf16* __restrict__ X,
                                                const __bf16* __restrict__ W1T,
                                                const float*  __restrict__ B1,
                                                __bf16* __restrict__ H) {
  __shared__ __bf16 As[2][256 * 40];   // 40 KB
  __shared__ __bf16 Bs[2][128 * 40];   // 20 KB

  const int tid  = threadIdx.x;
  const int lane = tid & 31;
  const int wave = tid >> 5;
  const int wm   = wave & 3;           // 4 strips of 64 rows
  const int wn   = wave >> 2;          // 2 strips of 64 cols
  const int m0   = blockIdx.y * 256;
  const int n0   = blockIdx.x * 128;

  // staging: A — thread t owns one row (64B = 4 chunks); B — half row (32B)
  const __bf16* gA = X + (size_t)(m0 + tid) * IN_DIM;
  const int brow = tid >> 1;
  const int bseg = (tid & 1) * 16;     // halves
  const __bf16* gB = W1T + (size_t)(n0 + brow) * IN_DIM + bseg;
  const int aLds = tid * 40;
  const int bLds = brow * 40 + bseg;

  // fragment addressing (16-bit A-matrix 16x32 layout, wave32)
  const int frow = lane & 15;
  const int fhi  = lane >> 4;          // K-half select
  const int fAoff = (wm * 64 + frow) * 40 + fhi * 16;
  const int fBoff = (wn * 64 + frow) * 40 + fhi * 16;

  const v8f vzero = {0.f, 0.f, 0.f, 0.f, 0.f, 0.f, 0.f, 0.f};
  v8f acc[4][4];
#pragma unroll
  for (int i = 0; i < 4; ++i)
#pragma unroll
    for (int j = 0; j < 4; ++j) acc[i][j] = vzero;

  // one K-step of fragment loads + 16 WMMAs from LDS buffer `cur`
  auto compute_step = [&](int cur) {
    V16 a[4], b[4];
#pragma unroll
    for (int ms = 0; ms < 4; ++ms) {
      const int4* p = (const int4*)&As[cur][fAoff + ms * 16 * 40];
      a[ms].i[0] = p[0]; a[ms].i[1] = p[1];
    }
#pragma unroll
    for (int ns = 0; ns < 4; ++ns) {
      const int4* p = (const int4*)&Bs[cur][fBoff + ns * 16 * 40];
      b[ns].i[0] = p[0]; b[ns].i[1] = p[1];
    }
#pragma unroll
    for (int ms = 0; ms < 4; ++ms)
#pragma unroll
      for (int ns = 0; ns < 4; ++ns)
        acc[ms][ns] = __builtin_amdgcn_wmma_f32_16x16x32_bf16(
            false, a[ms].v, false, b[ns].v, (short)0, acc[ms][ns],
            false, false);
  };

  const int KSTEPS = IN_DIM / 32;      // 96

  // prologue: stage K-step 0 into buffer 0
#pragma unroll
  for (int c = 0; c < 4; ++c) copy16_g2l(&As[0][aLds + c * 8], gA + c * 8);
#pragma unroll
  for (int c = 0; c < 2; ++c) copy16_g2l(&Bs[0][bLds + c * 8], gB + c * 8);
  wait_g2l();
  __syncthreads();

  // steady state: issue next tile's DMA, compute current, wait, barrier
#pragma unroll 1
  for (int step = 0; step < KSTEPS - 1; ++step) {
    const int cur = step & 1;
    const int nxt = cur ^ 1;
    const int k0  = (step + 1) * 32;
#pragma unroll
    for (int c = 0; c < 4; ++c)
      copy16_g2l(&As[nxt][aLds + c * 8], gA + k0 + c * 8);
#pragma unroll
    for (int c = 0; c < 2; ++c)
      copy16_g2l(&Bs[nxt][bLds + c * 8], gB + k0 + c * 8);

    compute_step(cur);

    wait_g2l();          // next buffer fully landed
    __syncthreads();     // before anyone reads it
  }

  // final K-step: no DMA, no trailing barrier
  compute_step((KSTEPS - 1) & 1);

  // epilogue: +bias, relu, bf16 store (lanes 0..15 = 16 consecutive n)
#pragma unroll
  for (int ns = 0; ns < 4; ++ns) {
    const int ng = n0 + wn * 64 + ns * 16 + frow;
    const float bias = B1[ng];
#pragma unroll
    for (int ms = 0; ms < 4; ++ms) {
#pragma unroll
      for (int r = 0; r < 8; ++r) {
        const int mg = m0 + wm * 64 + ms * 16 + r + fhi * 8;
        float v = acc[ms][ns][r] + bias;
        v = v > 0.f ? v : 0.f;
        H[(size_t)mg * HIDDEN + ng] = (__bf16)v;
      }
    }
  }
}

// ---------------- fc2: [8192,4096] x [4096,16(pad)] + bias ----------------
// One wave per 16-row output tile; streaming WMMA over K=4096 (L2-resident).

__global__ __launch_bounds__(256) void fc2_gemm(const __bf16* __restrict__ H,
                                                const __bf16* __restrict__ W2T,
                                                const float*  __restrict__ B2,
                                                float* __restrict__ OUT) {
  const int tid  = threadIdx.x;
  const int lane = tid & 31;
  const int wave = tid >> 5;
  const int m0   = (blockIdx.x * 8 + wave) * 16;
  const int frow = lane & 15;
  const int fhi  = lane >> 4;

  const int4* gA = (const int4*)(H   + (size_t)(m0 + frow) * HIDDEN + fhi * 16);
  const int4* gB = (const int4*)(W2T + (size_t)frow        * HIDDEN + fhi * 16);

  v8f acc = {0.f, 0.f, 0.f, 0.f, 0.f, 0.f, 0.f, 0.f};
  const int KSTEPS = HIDDEN / 32;          // 128
  for (int step = 0; step < KSTEPS; ++step) {
    V16 a, b;
    const int o = step * 4;
    a.i[0] = gA[o]; a.i[1] = gA[o + 1];
    b.i[0] = gB[o]; b.i[1] = gB[o + 1];
    acc = __builtin_amdgcn_wmma_f32_16x16x32_bf16(false, a.v, false, b.v,
                                                  (short)0, acc, false, false);
  }

  if (frow < OUT_DIM) {                    // diverge only after all WMMAs
    const float bias = B2[frow];
#pragma unroll
    for (int r = 0; r < 8; ++r) {
      const int mg = m0 + r + fhi * 8;
      OUT[(size_t)mg * OUT_DIM + frow] = acc[r] + bias;
    }
  }
}

// ---------------- launch ---------------------------------------------------

extern "C" void kernel_launch(void* const* d_in, const int* in_sizes, int n_in,
                              void* d_out, int out_size, void* d_ws, size_t ws_size,
                              hipStream_t stream) {
  const float* x  = (const float*)d_in[0];
  const float* W1 = (const float*)d_in[1];
  const float* b1 = (const float*)d_in[2];
  const float* W2 = (const float*)d_in[3];
  const float* b2 = (const float*)d_in[4];
  float* out = (float*)d_out;

  char* ws = (char*)d_ws;
  const size_t XB_BYTES  = (size_t)BATCH  * IN_DIM * 2;   // 50.3 MB
  const size_t W1T_BYTES = (size_t)HIDDEN * IN_DIM * 2;   // 25.2 MB
  const size_t W2T_BYTES = (size_t)16 * HIDDEN * 2;       // 128 KB
  __bf16* xb  = (__bf16*)ws;
  __bf16* w1t = (__bf16*)(ws + XB_BYTES);
  __bf16* w2t = (__bf16*)(ws + XB_BYTES + W1T_BYTES);
  __bf16* h   = (__bf16*)(ws + XB_BYTES + W1T_BYTES + W2T_BYTES); // 67.1 MB

  // 1) x fp32 -> bf16
  {
    int n4 = (BATCH * IN_DIM) / 4;
    cvt_f32_bf16x4<<<(n4 + 255) / 256, 256, 0, stream>>>(x, xb, n4);
  }
  // 2) W1 -> W1^T bf16
  {
    dim3 g(HIDDEN / 32, IN_DIM / 32);
    w1_transpose<<<g, 256, 0, stream>>>(W1, w1t);
  }
  // 3) W2 -> padded W2^T bf16
  w2_prep<<<(16 * HIDDEN) / 256, 256, 0, stream>>>(W2, w2t);

  // 4) fc1 (bias + relu fused)
  {
    dim3 g(HIDDEN / 128, BATCH / 256);
    fc1_gemm<<<g, 256, 0, stream>>>(xb, w1t, b1, h);
  }
  // 5) fc2 (bias fused)
  fc2_gemm<<<BATCH / (16 * 8), 256, 0, stream>>>(h, w2t, b2, out);
}
